// LlamaAttention_77996606095760
// MI455X (gfx1250) — compile-verified, matrix operand order checked by reading, not run on hardware
//
#include <hip/hip_runtime.h>

#define S_LEN 2048
#define HID   4096
#define NH    32
#define NKV   8
#define HD    128
#define QKV_N ((NH + 2 * NKV) * HD)          // 6144
#define ATT_SCALE 0.08838834764831845f       // 128^-0.5

typedef __attribute__((ext_vector_type(16))) __bf16 v16bf;
typedef __attribute__((ext_vector_type(8)))  __bf16 v8bf;
typedef __attribute__((ext_vector_type(8)))  float  v8f;
typedef __attribute__((ext_vector_type(4)))  int    v4i;

#if defined(__has_builtin)
#if __has_builtin(__builtin_amdgcn_global_load_async_to_lds_b128)
#define HAVE_ASYNC_LDS 1
#endif
#endif

// round-to-nearest-even f32 -> bf16
static __device__ __forceinline__ __bf16 to_bf16(float f) {
  unsigned u = __builtin_bit_cast(unsigned, f);
  u += 0x7FFFu + ((u >> 16) & 1u);
  unsigned short h = (unsigned short)(u >> 16);
  return __builtin_bit_cast(__bf16, h);
}

static __device__ __forceinline__ v8f wmma_bf16(v16bf a, v16bf b, v8f c) {
  return __builtin_amdgcn_wmma_f32_16x16x32_bf16(
      /*neg_a=*/false, a, /*neg_b=*/false, b,
      /*c_mod=*/(short)0, c, /*reuse_a=*/false, /*reuse_b=*/false);
}

// Load a 16x32 (MxK) WMMA operand fragment from row-major [row][col] storage.
// Per ISA layout: lanes 0-15 hold row (base+lane), K = col0+0..7 and col0+16..23;
// lanes 16-31 hold the same rows, K = col0+8..15 and col0+24..31.
// Works for A fragments and (since B of C=A*B^T is row-major [n][k]) B fragments.
static __device__ __forceinline__ v16bf load_frag(const __bf16* base, int row0,
                                                  int col0, int pitch) {
  const int lane = threadIdx.x & 31;
  const int half = lane >> 4;
  const __bf16* p = base + (size_t)(row0 + (lane & 15)) * pitch + col0 + half * 8;
  v8bf lo = *(const v8bf*)p;         // K +0..7  (or +8..15)
  v8bf hi = *(const v8bf*)(p + 16);  // K +16..23 (or +24..31)
  return __builtin_shufflevector(lo, hi, 0, 1, 2, 3, 4, 5, 6, 7,
                                 8, 9, 10, 11, 12, 13, 14, 15);
}

// ---------------------------------------------------------------------------
// GEMM: C[M,N] = A[M,K] * B[N,K]^T   (A,B fp32 row-major; bf16 WMMA inside)
// grid = (N/128, M/128), block = 256 (8 waves: 2 (M) x 4 (N)).
// BK=64, register double-buffered: next global tile loads overlap WMMA work.
// ---------------------------------------------------------------------------
#define BM 128
#define BN 128
#define BK 64
#define KP 72  // LDS pitch (bf16): 144B rows -> 16B aligned, conflict-free

__global__ __launch_bounds__(256) void gemm_bf16(const float* __restrict__ A,
                                                 const float* __restrict__ B,
                                                 float* __restrict__ C,
                                                 int K, int N) {
  __shared__ __bf16 As[BM * KP];
  __shared__ __bf16 Bs[BN * KP];

  const int m0 = blockIdx.y * BM;
  const int n0 = blockIdx.x * BN;
  const int t = threadIdx.x;
  const int wave = t >> 5;
  const int wm = wave >> 2;  // 0..1  -> 64 rows
  const int wn = wave & 3;   // 0..3  -> 32 cols
  const int lane = t & 31;
  const int half = lane >> 4;
  const int ln = lane & 15;

  // staging: each thread moves 32 floats per matrix (half a 64-col row)
  const int sr = t >> 1;
  const int stc = (t & 1) * 32;

  const float* pA = A + (size_t)(m0 + sr) * K + stc;
  const float* pB = B + (size_t)(n0 + sr) * K + stc;

  float4 ra[8], rb[8];
#define LDG(dst, src)                                            \
  {                                                              \
    const float4* g_ = (const float4*)(src);                     \
    _Pragma("unroll") for (int j = 0; j < 8; ++j) dst[j] = g_[j];\
  }
#define STS(src, smem)                                                        \
  {                                                                           \
    __bf16* s_ = (smem) + sr * KP + stc;                                      \
    _Pragma("unroll") for (int j = 0; j < 4; ++j) {                           \
      float4 e_ = src[2 * j], f_ = src[2 * j + 1];                            \
      v8bf p_ = {to_bf16(e_.x), to_bf16(e_.y), to_bf16(e_.z), to_bf16(e_.w), \
                 to_bf16(f_.x), to_bf16(f_.y), to_bf16(f_.z), to_bf16(f_.w)};\
      *(v8bf*)&s_[j * 8] = p_;                                                \
    }                                                                         \
  }

  v8f acc[4][2] = {};

  LDG(ra, pA)
  LDG(rb, pB)
  STS(ra, As)
  STS(rb, Bs)
  __syncthreads();

  const int nk = K / BK;
  for (int kt = 0; kt < nk; ++kt) {
    if (kt + 1 < nk) {
      LDG(ra, pA + (size_t)(kt + 1) * BK)
      LDG(rb, pB + (size_t)(kt + 1) * BK)
      if (kt + 2 < nk) {
        __builtin_prefetch(pA + (size_t)(kt + 2) * BK, 0, 1);
        __builtin_prefetch(pB + (size_t)(kt + 2) * BK, 0, 1);
      }
    }
#pragma unroll
    for (int ks = 0; ks < 2; ++ks) {
      v16bf bf0 = load_frag(Bs, wn * 32, ks * 32, KP);
      v16bf bf1 = load_frag(Bs, wn * 32 + 16, ks * 32, KP);
#pragma unroll
      for (int mi = 0; mi < 4; ++mi) {
        v16bf af = load_frag(As, wm * 64 + mi * 16, ks * 32, KP);
        acc[mi][0] = wmma_bf16(af, bf0, acc[mi][0]);
        acc[mi][1] = wmma_bf16(af, bf1, acc[mi][1]);
      }
    }
    __syncthreads();
    if (kt + 1 < nk) {
      STS(ra, As)
      STS(rb, Bs)
      __syncthreads();
    }
  }

  // C/D layout: VGPR g -> row g + 8*half, lane -> column ln
#pragma unroll
  for (int mi = 0; mi < 4; ++mi)
#pragma unroll
    for (int ni = 0; ni < 2; ++ni)
#pragma unroll
      for (int g = 0; g < 8; ++g) {
        int row = m0 + wm * 64 + mi * 16 + g + 8 * half;
        int col = n0 + wn * 32 + ni * 16 + ln;
        C[(size_t)row * N + col] = acc[mi][ni][g];
      }
}

// ---------------------------------------------------------------------------
// RoPE + bf16 conversion kernels
// ---------------------------------------------------------------------------
__global__ void rope_q_kernel(const float* __restrict__ qkv,
                              const int* __restrict__ pos,
                              __bf16* __restrict__ qo) {
  int tid = blockIdx.x * blockDim.x + threadIdx.x;  // S*NH*64
  int i = tid & 63;
  int h = (tid >> 6) & (NH - 1);
  int s = tid >> 11;
  const float* row = qkv + (size_t)s * QKV_N + h * HD;
  float x1 = row[i], x2 = row[i + 64];
  float fr = (float)pos[s] * __powf(10000.f, -(float)i * (1.f / 64.f));
  float sn, cs;
  __sincosf(fr, &sn, &cs);
  __bf16* orow = qo + (size_t)s * (NH * HD) + h * HD;
  orow[i] = to_bf16(x1 * cs - x2 * sn);
  orow[i + 64] = to_bf16(x2 * cs + x1 * sn);
}

__global__ void rope_k_kernel(const float* __restrict__ qkv,
                              const int* __restrict__ pos,
                              __bf16* __restrict__ ko) {
  int tid = blockIdx.x * blockDim.x + threadIdx.x;  // S*NKV*64
  int i = tid & 63;
  int h = (tid >> 6) & (NKV - 1);
  int s = tid >> 9;
  const float* row = qkv + (size_t)s * QKV_N + NH * HD + h * HD;
  float x1 = row[i], x2 = row[i + 64];
  float fr = (float)pos[s] * __powf(10000.f, -(float)i * (1.f / 64.f));
  float sn, cs;
  __sincosf(fr, &sn, &cs);
  __bf16* orow = ko + (size_t)s * (NKV * HD) + h * HD;
  orow[i] = to_bf16(x1 * cs - x2 * sn);
  orow[i + 64] = to_bf16(x2 * cs + x1 * sn);
}

__global__ void conv_v_kernel(const float* __restrict__ qkv,
                              __bf16* __restrict__ vo) {
  int tid = blockIdx.x * blockDim.x + threadIdx.x;  // S*NKV*HD
  int j = tid & (NKV * HD - 1);
  int s = tid >> 10;
  vo[(size_t)s * (NKV * HD) + j] =
      to_bf16(qkv[(size_t)s * QKV_N + (NH + NKV) * HD + j]);
}

// ---------------------------------------------------------------------------
// Flash attention fwd (GQA, causal). grid = (S/128, NH), block = 256 (8 waves,
// each wave owns 16 q-rows). 64-key k-blocks; V staged transposed in LDS.
// K tile staged with CDNA5 async-to-LDS when the builtin is available.
// ---------------------------------------------------------------------------
#define BQ 128
#define BKV 64
#define KSP 136  // Ks pitch over d (272B rows)
#define VSP 72   // Vt / Pw pitch over key-pos (144B rows)

__global__ __launch_bounds__(256) void attn_fwd(const __bf16* __restrict__ Q,
                                                const __bf16* __restrict__ Kc,
                                                const __bf16* __restrict__ Vc,
                                                float* __restrict__ O) {
  __shared__ __bf16 Ks[BKV * KSP];     // [key pos][d]
  __shared__ __bf16 Vt[HD * VSP];      // [d][key pos]   (transposed)
  __shared__ __bf16 Pw[8][16 * VSP];   // per-wave probs [q row][key pos]

  const int qb = blockIdx.x;
  const int h = blockIdx.y;
  const int kvh = h >> 2;  // G = NH/NKV = 4
  const int t = threadIdx.x;
  const int wave = t >> 5;
  const int lane = t & 31;
  const int half = lane >> 4;
  const int ln = lane & 15;
  const int sbase = qb * BQ + wave * 16;

  // Q fragments (16 rows x 128 d, 4 chunks of K=32) straight from global
  v16bf qf[4];
#pragma unroll
  for (int d = 0; d < 4; ++d)
    qf[d] = load_frag(Q + (size_t)sbase * (NH * HD) + h * HD, 0, d * 32, NH * HD);

  v8f o[8] = {};
  float m_s[8], l_s[8];
#pragma unroll
  for (int g = 0; g < 8; ++g) { m_s[g] = -1e30f; l_s[g] = 0.f; }

  const int kb_end = 2 * qb + 1;
  for (int kb = 0; kb <= kb_end; ++kb) {
    __syncthreads();
    // stage K tile (64 x 128) and V tile transposed (128 d x 64 pos)
    {
      int r = t >> 2;
      int c = (t & 3) * 32;
      const __bf16* gk =
          Kc + (size_t)(kb * BKV + r) * (NKV * HD) + kvh * HD + c;
#if defined(HAVE_ASYNC_LDS)
#pragma unroll
      for (int j = 0; j < 4; ++j)
        __builtin_amdgcn_global_load_async_to_lds_b128(
            (__attribute__((address_space(1))) v4i*)(gk + j * 8),
            (__attribute__((address_space(3))) v4i*)&Ks[r * KSP + c + j * 8],
            0, 0);
#else
      const v8bf* gk4 = (const v8bf*)gk;
#pragma unroll
      for (int j = 0; j < 4; ++j) *(v8bf*)&Ks[r * KSP + c + j * 8] = gk4[j];
#endif
      const v8bf* gv = (const v8bf*)(Vc + (size_t)(kb * BKV + r) * (NKV * HD) +
                                     kvh * HD + c);
      v8bf vv[4];
#pragma unroll
      for (int j = 0; j < 4; ++j) vv[j] = gv[j];
#pragma unroll
      for (int j = 0; j < 4; ++j)
#pragma unroll
        for (int e = 0; e < 8; ++e)
          Vt[(c + j * 8 + e) * VSP + r] = vv[j][e];
    }
#if defined(HAVE_ASYNC_LDS)
#if __has_builtin(__builtin_amdgcn_s_wait_asynccnt)
    __builtin_amdgcn_s_wait_asynccnt(0);
#else
    asm volatile("s_wait_asynccnt 0x0" ::: "memory");
#endif
#endif
    __syncthreads();

    // scores: S = Q * K^T  (16 x 64 per wave)
    v8f sc[4] = {};
#pragma unroll
    for (int d = 0; d < 4; ++d)
#pragma unroll
      for (int nt = 0; nt < 4; ++nt)
        sc[nt] = wmma_bf16(qf[d], load_frag(Ks, nt * 16, d * 32, KSP), sc[nt]);

    // scale + causal mask (only blocks touching the diagonal)
    const bool diag = (kb >= 2 * qb);
#pragma unroll
    for (int nt = 0; nt < 4; ++nt)
#pragma unroll
      for (int g = 0; g < 8; ++g) {
        float v = sc[nt][g] * ATT_SCALE;
        if (diag) {
          int row = sbase + g + 8 * half;
          int col = kb * BKV + nt * 16 + ln;
          if (col > row) v = -1e30f;
        }
        sc[nt][g] = v;
      }

    // online softmax (row stats live per lane-half; reduce across 16 lanes)
    float sf[8];
#pragma unroll
    for (int g = 0; g < 8; ++g) {
      float mx = m_s[g];
#pragma unroll
      for (int nt = 0; nt < 4; ++nt) mx = fmaxf(mx, sc[nt][g]);
      for (int d = 8; d >= 1; d >>= 1) mx = fmaxf(mx, __shfl_xor(mx, d, 32));
      sf[g] = __expf(m_s[g] - mx);
      m_s[g] = mx;
      float rs = 0.f;
#pragma unroll
      for (int nt = 0; nt < 4; ++nt) {
        float p = __expf(sc[nt][g] - mx);
        sc[nt][g] = p;
        rs += p;
      }
      for (int d = 8; d >= 1; d >>= 1) rs += __shfl_xor(rs, d, 32);
      l_s[g] = l_s[g] * sf[g] + rs;
    }
#pragma unroll
    for (int dt = 0; dt < 8; ++dt)
#pragma unroll
      for (int g = 0; g < 8; ++g) o[dt][g] *= sf[g];

    // write P (bf16) to this wave's LDS scratch, then feed it back as A frags
    __bf16* pw = &Pw[wave][0];
#pragma unroll
    for (int nt = 0; nt < 4; ++nt)
#pragma unroll
      for (int g = 0; g < 8; ++g)
        pw[(g + 8 * half) * VSP + nt * 16 + ln] = to_bf16(sc[nt][g]);
    asm volatile("s_wait_dscnt 0" ::: "memory");  // same-wave LDS RAW

    // O += P * V
#pragma unroll
    for (int ks = 0; ks < 2; ++ks) {
      v16bf pf = load_frag(pw, 0, ks * 32, VSP);
#pragma unroll
      for (int dt = 0; dt < 8; ++dt)
        o[dt] = wmma_bf16(pf, load_frag(Vt, dt * 16, ks * 32, VSP), o[dt]);
    }
  }

  // epilogue: normalize and store [S, NH*D]
#pragma unroll
  for (int g = 0; g < 8; ++g) {
    float inv = 1.0f / l_s[g];
    int row = sbase + g + 8 * half;
#pragma unroll
    for (int dt = 0; dt < 8; ++dt)
      O[(size_t)row * (NH * HD) + h * HD + dt * 16 + ln] = o[dt][g] * inv;
  }
}

// ---------------------------------------------------------------------------
extern "C" void kernel_launch(void* const* d_in, const int* in_sizes, int n_in,
                              void* d_out, int out_size, void* d_ws,
                              size_t ws_size, hipStream_t stream) {
  (void)in_sizes; (void)n_in; (void)out_size; (void)ws_size;
  const float* hidden = (const float*)d_in[0];     // [S, H]
  const int* positions = (const int*)d_in[1];      // [S]
  const float* w_qkv = (const float*)d_in[2];      // [6144, 4096] row-major
  const float* w_o = (const float*)d_in[3];        // [H, NH*D]   row-major
  float* out = (float*)d_out;                      // [S, H]

  char* w = (char*)d_ws;
  float* qkv = (float*)w;    w += (size_t)S_LEN * QKV_N * sizeof(float);
  __bf16* qb16 = (__bf16*)w; w += (size_t)S_LEN * NH * HD * sizeof(__bf16);
  __bf16* kb16 = (__bf16*)w; w += (size_t)S_LEN * NKV * HD * sizeof(__bf16);
  __bf16* vb16 = (__bf16*)w; w += (size_t)S_LEN * NKV * HD * sizeof(__bf16);
  float* aout = (float*)w;   // [S, NH*D]

  // 1) fused QKV projection: qkv = hidden @ w_qkv^T
  gemm_bf16<<<dim3(QKV_N / BN, S_LEN / BM), 256, 0, stream>>>(hidden, w_qkv,
                                                              qkv, HID, QKV_N);
  // 2) RoPE + bf16 conversion
  rope_q_kernel<<<(S_LEN * NH * 64) / 256, 256, 0, stream>>>(qkv, positions, qb16);
  rope_k_kernel<<<(S_LEN * NKV * 64) / 256, 256, 0, stream>>>(qkv, positions, kb16);
  conv_v_kernel<<<(S_LEN * NKV * HD) / 256, 256, 0, stream>>>(qkv, vb16);
  // 3) causal GQA flash attention
  attn_fwd<<<dim3(S_LEN / BQ, NH), 256, 0, stream>>>(qb16, kb16, vb16, aout);
  // 4) output projection: out = aout @ w_o^T
  gemm_bf16<<<dim3(HID / BN, S_LEN / BM), 256, 0, stream>>>(aout, w_o, out,
                                                            NH * HD, HID);
}
